// MultilayerGRU_738734375312
// MI455X (gfx1250) — matrix operand order repaired and checked by reading, not compiled, not run on hardware
//
#include <hip/hip_runtime.h>
#include <hip/hip_bf16.h>

// ---------------------------------------------------------------------------
// Multilayer GRU on gfx1250: bf16 WMMA (V_WMMA_F32_16X16X32_BF16) everywhere.
// B=32, S=512, I=H=O=1024, L=3.
// Recurrent scan uses split-K(4) per tile + LDS reduction to shorten the
// dependent WMMA chain (latency-bound; weights are L2-resident).
// ---------------------------------------------------------------------------

typedef __attribute__((ext_vector_type(16))) __bf16 v16bf;
typedef __attribute__((ext_vector_type(8)))  float  v8f;

#define GRU_B 32
#define GRU_S 512
#define GRU_I 1024
#define GRU_H 1024
#define GRU_O 1024
#define GRU_L 3

union FragBF { uint4 u[2]; v16bf v; };

__device__ inline v8f wmma_bf16(v16bf a, v16bf b, v8f c) {
  return __builtin_amdgcn_wmma_f32_16x16x32_bf16(false, a, false, b,
                                                 (short)0, c, false, false);
}

// A-fragment (16x32, bf16, row-major source): lane holds row m=lane%16,
// VGPRs 0-3 = K[half*8 .. +8), VGPRs 4-7 = K[16+half*8 .. +8).
__device__ inline v16bf load_frag_A_bf16(const __bf16* base, int lda, int kc) {
  const int lane = threadIdx.x & 31;
  const int half = lane >> 4;
  const int m    = lane & 15;
  const __bf16* p = base + (size_t)m * lda + kc + half * 8;
  FragBF f;
  f.u[0] = *reinterpret_cast<const uint4*>(p);       // K chunk 0 (8 bf16)
  f.u[1] = *reinterpret_cast<const uint4*>(p + 16);  // K chunk 1 (8 bf16)
  return f.v;
}

// Same A-fragment but converting from fp32 source on the fly (layer 0 reads x).
__device__ inline v16bf load_frag_A_f32(const float* base, int lda, int kc) {
  const int lane = threadIdx.x & 31;
  const int half = lane >> 4;
  const int m    = lane & 15;
  const float* p = base + (size_t)m * lda + kc + half * 8;
  v16bf r;
#pragma unroll
  for (int i = 0; i < 8; ++i) r[i]     = (__bf16)p[i];
#pragma unroll
  for (int i = 0; i < 8; ++i) r[8 + i] = (__bf16)p[16 + i];
  return r;
}

// B-fragment (32x16): out = inp * W^T so B[k][n] = W[n][k]; lane n=lane%16
// holds W[n0+n][kc + half*8 .. +8) and W[n0+n][kc + 16 + half*8 .. +8), row-major W.
__device__ inline v16bf load_frag_B(const __bf16* W, int ldw, int n0, int kc) {
  const int lane = threadIdx.x & 31;
  const int half = lane >> 4;
  const int n    = lane & 15;
  const __bf16* p = W + (size_t)(n0 + n) * ldw + kc + half * 8;
  FragBF f;
  f.u[0] = *reinterpret_cast<const uint4*>(p);
  f.u[1] = *reinterpret_cast<const uint4*>(p + 16);
  return f.v;
}

__device__ inline v8f zero_v8f() {
  v8f z;
#pragma unroll
  for (int i = 0; i < 8; ++i) z[i] = 0.0f;
  return z;
}

// ---------------------------------------------------------------------------
// fp32 -> bf16 weight conversion (weights stay row-major; fragment loads are
// contiguous by construction).
// ---------------------------------------------------------------------------
__global__ void convert_f32_bf16(const float* __restrict__ src,
                                 __bf16* __restrict__ dst, int n) {
  int i = blockIdx.x * 256 + threadIdx.x;
  if (i < n) dst[i] = (__bf16)src[i];
}

// h0 (B,L,H) -> per-layer state hf[(l*B+b)*H+h] (f32 + bf16 mirror)
__global__ void init_h(const float* __restrict__ h0,
                       float* __restrict__ hf, __bf16* __restrict__ hbf) {
  int i = blockIdx.x * 256 + threadIdx.x;            // i < B*L*H
  if (i >= GRU_B * GRU_L * GRU_H) return;
  int h = i & (GRU_H - 1);
  int l = (i >> 10) % GRU_L;
  int b = i / (GRU_L * GRU_H);
  float v = h0[i];
  int j = (l * GRU_B + b) * GRU_H + h;
  hf[j]  = v;
  hbf[j] = (__bf16)v;
}

// final hf[(l*B+b)*H+h] -> d_out hidden_state (B,L,H)
__global__ void copy_hidden(const float* __restrict__ hf,
                            float* __restrict__ out) {
  int i = blockIdx.x * 256 + threadIdx.x;
  if (i >= GRU_B * GRU_L * GRU_H) return;
  int h = i & (GRU_H - 1);
  int l = (i >> 10) % GRU_L;
  int b = i / (GRU_L * GRU_H);
  out[i] = hf[(l * GRU_B + b) * GRU_H + h];
}

// ---------------------------------------------------------------------------
// One GRU layer for one timestep.
// Grid: 128 blocks (one per 16x16 output tile: nt 0..63 x mt 0..1),
// 128 threads = 4 waves per block. Each wave accumulates a K/4 slice of both
// the input and hidden projections (48 WMMAs), then waves 1-3 spill partials
// to LDS and wave 0 reduces + applies the gate nonlinearity + state update.
// Accumulators: z, r, g_input, g_hidden kept separate (r gates only the
// hidden part of g).
// ---------------------------------------------------------------------------
template <bool A_F32>
__global__ __launch_bounds__(128) void gru_layer_step(
    const float*  __restrict__ xA_f32,   // layer-0 input rows (row stride ldx)
    const __bf16* __restrict__ xA_bf,    // layer>0 input rows (row stride ldx)
    int ldx, int K_in,
    const __bf16* __restrict__ Wzi, const __bf16* __restrict__ Wri,
    const __bf16* __restrict__ Wgi,                         // (H x K_in)
    const __bf16* __restrict__ Wzh, const __bf16* __restrict__ Wrh,
    const __bf16* __restrict__ Wgh,                         // (H x H)
    const float* __restrict__ bz, const float* __restrict__ br,
    const float* __restrict__ bg,                           // (H)
    const float*  __restrict__ h_in_f32,                    // (B x H) old state
    const __bf16* __restrict__ h_in_bf,                     // (B x H) old state
    float*  __restrict__ h_out_f32,                         // (B x H) new state
    __bf16* __restrict__ h_out_bf,                          // (B x H) new state
    __bf16* __restrict__ hist)                              // optional history
{
  __shared__ float red[3][4][256];       // 12 KB: waves 1-3 x 4 accs x 256

  const int wave = threadIdx.x >> 5;     // 0..3 = K-slice
  const int lane = threadIdx.x & 31;
  const int tile = blockIdx.x;           // 0..127
  const int nt = tile >> 1;
  const int mt = tile & 1;
  const int n0 = nt * 16;

  v8f acc[4];                            // 0:z 1:r 2:g_in 3:g_hid
#pragma unroll
  for (int g = 0; g < 4; ++g) acc[g] = zero_v8f();

  // ---- input projection K-slice: inp @ {Wzi,Wri,Wgi}^T ----
  {
    const int ks = (K_in >> 2) * wave;
    const int ke = ks + (K_in >> 2);
    for (int kc = ks; kc < ke; kc += 32) {
      v16bf a;
      if constexpr (A_F32)
        a = load_frag_A_f32(xA_f32 + (size_t)(mt * 16) * ldx, ldx, kc);
      else
        a = load_frag_A_bf16(xA_bf + (size_t)(mt * 16) * ldx, ldx, kc);
      acc[0] = wmma_bf16(a, load_frag_B(Wzi, K_in, n0, kc), acc[0]);
      acc[1] = wmma_bf16(a, load_frag_B(Wri, K_in, n0, kc), acc[1]);
      acc[2] = wmma_bf16(a, load_frag_B(Wgi, K_in, n0, kc), acc[2]);
    }
  }

  // ---- hidden projection K-slice: h @ {Wzh,Wrh,Wgh}^T ----
  {
    const int ks = (GRU_H >> 2) * wave;
    const int ke = ks + (GRU_H >> 2);
    for (int kc = ks; kc < ke; kc += 32) {
      v16bf a =
          load_frag_A_bf16(h_in_bf + (size_t)(mt * 16) * GRU_H, GRU_H, kc);
      acc[0] = wmma_bf16(a, load_frag_B(Wzh, GRU_H, n0, kc), acc[0]);
      acc[1] = wmma_bf16(a, load_frag_B(Wrh, GRU_H, n0, kc), acc[1]);
      acc[3] = wmma_bf16(a, load_frag_B(Wgh, GRU_H, n0, kc), acc[3]);
    }
  }

  // ---- split-K reduction in LDS ----
  if (wave > 0) {
#pragma unroll
    for (int g = 0; g < 4; ++g)
#pragma unroll
      for (int v = 0; v < 8; ++v) red[wave - 1][g][v * 32 + lane] = acc[g][v];
  }
  __syncthreads();
  if (wave != 0) return;

#pragma unroll
  for (int w = 0; w < 3; ++w)
#pragma unroll
    for (int g = 0; g < 4; ++g)
#pragma unroll
      for (int v = 0; v < 8; ++v) acc[g][v] += red[w][g][v * 32 + lane];

  // ---- gates + state update (C/D layout: m = v + 8*(lane>=16), n = lane%16)
  const int n     = n0 + (lane & 15);
  const int mbase = mt * 16 + ((lane >> 4) ? 8 : 0);
  const float vbz = bz[n], vbr = br[n], vbg = bg[n];

#pragma unroll
  for (int v = 0; v < 8; ++v) {
    const int m = mbase + v;
    float z = 1.0f / (1.0f + __expf(-(acc[0][v] + vbz)));
    float r = 1.0f / (1.0f + __expf(-(acc[1][v] + vbr)));
    float g = tanhf(acc[2][v] + r * (acc[3][v] + vbg));
    float hold = h_in_f32[m * GRU_H + n];
    float hn = z * hold + (1.0f - z) * g;
    h_out_f32[m * GRU_H + n] = hn;
    __bf16 hb = (__bf16)hn;
    h_out_bf[m * GRU_H + n] = hb;
    if (hist) hist[m * GRU_H + n] = hb;
  }
}

// ---------------------------------------------------------------------------
// Output projection: out[b][t][o] = hist[t][b] @ Wout^T + bout.
// M = S*B = 16384 rows (1024 m-tiles), N = 1024 (64 n-tiles) -> 65536 tiles.
// ---------------------------------------------------------------------------
__global__ __launch_bounds__(256) void out_gemm(
    const __bf16* __restrict__ hist, const __bf16* __restrict__ Wout,
    const float* __restrict__ bout, float* __restrict__ out) {
  const int wave  = threadIdx.x >> 5;
  const int tile  = blockIdx.x * 8 + wave;     // 8192 blocks * 8 waves
  const int nt    = tile & 63;
  const int mtile = tile >> 6;                 // 0..1023
  const int t     = mtile >> 1;
  const int b0    = (mtile & 1) * 16;

  const __bf16* A = hist + (size_t)(t * GRU_B + b0) * GRU_H;

  v8f acc = zero_v8f();
  for (int kc = 0; kc < GRU_H; kc += 32)
    acc = wmma_bf16(load_frag_A_bf16(A, GRU_H, kc),
                    load_frag_B(Wout, GRU_H, nt * 16, kc), acc);

  const int lane = threadIdx.x & 31;
  const int n  = nt * 16 + (lane & 15);
  const int mb = b0 + ((lane >> 4) ? 8 : 0);
  const float bb = bout[n];
#pragma unroll
  for (int v = 0; v < 8; ++v) {
    const int b = mb + v;
    out[((size_t)b * GRU_S + t) * GRU_O + n] = acc[v] + bb;
  }
}

// ---------------------------------------------------------------------------
extern "C" void kernel_launch(void* const* d_in, const int* in_sizes, int n_in,
                              void* d_out, int out_size, void* d_ws,
                              size_t ws_size, hipStream_t stream) {
  const float* x    = (const float*)d_in[0];   // (B,S,I)
  const float* h0   = (const float*)d_in[1];   // (B,L,H)
  const float* Wzi  = (const float*)d_in[2];   // (L,H,I)
  const float* Wzh  = (const float*)d_in[3];   // (L,H,H)
  const float* bzh  = (const float*)d_in[4];   // (L,H)
  const float* Wri  = (const float*)d_in[5];
  const float* Wrh  = (const float*)d_in[6];
  const float* brh  = (const float*)d_in[7];
  const float* Wgi  = (const float*)d_in[8];
  const float* Wgh  = (const float*)d_in[9];
  const float* bgh  = (const float*)d_in[10];
  const float* Wout = (const float*)d_in[11];  // (O,H)
  const float* bout = (const float*)d_in[12];  // (O)

  float* out_seq = (float*)d_out;                                  // (B,S,O)
  float* out_hid = out_seq + (size_t)GRU_B * GRU_S * GRU_O;        // (B,L,H)

  // ---- workspace layout (~71 MB) ----
  const size_t HI = (size_t)GRU_H * GRU_I;       // 1M elements
  const size_t BH = (size_t)GRU_B * GRU_H;       // 32K elements
  __bf16* wzi  = (__bf16*)d_ws;                  // 3*HI
  __bf16* wzh  = wzi + 3 * HI;
  __bf16* wri  = wzh + 3 * HI;
  __bf16* wrh  = wri + 3 * HI;
  __bf16* wgi  = wrh + 3 * HI;
  __bf16* wgh  = wgi + 3 * HI;
  __bf16* wout = wgh + 3 * HI;                   // HI
  __bf16* hist = wout + HI;                      // S*B*H = 16M
  float*  hf[2];
  hf[0] = (float*)(hist + (size_t)GRU_S * BH);
  hf[1] = hf[0] + GRU_L * BH;
  __bf16* hbf[2];
  hbf[0] = (__bf16*)(hf[1] + GRU_L * BH);
  hbf[1] = hbf[0] + GRU_L * BH;

  // ---- weights fp32 -> bf16 (row-major preserved) ----
  auto cvt = [&](const float* s, __bf16* d, int n) {
    convert_f32_bf16<<<(n + 255) / 256, 256, 0, stream>>>(s, d, n);
  };
  cvt(Wzi, wzi, (int)(3 * HI));
  cvt(Wzh, wzh, (int)(3 * HI));
  cvt(Wri, wri, (int)(3 * HI));
  cvt(Wrh, wrh, (int)(3 * HI));
  cvt(Wgi, wgi, (int)(3 * HI));
  cvt(Wgh, wgh, (int)(3 * HI));
  cvt(Wout, wout, (int)HI);

  // ---- initial hidden state ----
  init_h<<<(GRU_B * GRU_L * GRU_H + 255) / 256, 256, 0, stream>>>(
      h0, hf[0], hbf[0]);

  // ---- sequential scan: ping-pong h buffers; layer l+1 reads layer l's
  //      just-written bf16 hidden (stream order = sync) ----
  for (int t = 0; t < GRU_S; ++t) {
    const int rb = t & 1, wb = (t + 1) & 1;
    // layer 0: input A comes from x (fp32, row stride S*I)
    gru_layer_step<true><<<128, 128, 0, stream>>>(
        x + (size_t)t * GRU_I, nullptr, GRU_S * GRU_I, GRU_I,
        wzi, wri, wgi, wzh, wrh, wgh,
        bzh, brh, bgh,
        hf[rb], hbf[rb], hf[wb], hbf[wb], nullptr);
    // layer 1: input = layer-0 hidden just written this step
    gru_layer_step<false><<<128, 128, 0, stream>>>(
        nullptr, hbf[wb], GRU_H, GRU_H,
        wzi + HI, wri + HI, wgi + HI, wzh + HI, wrh + HI, wgh + HI,
        bzh + GRU_H, brh + GRU_H, bgh + GRU_H,
        hf[rb] + BH, hbf[rb] + BH, hf[wb] + BH, hbf[wb] + BH, nullptr);
    // layer 2: input = layer-1 hidden; also record bf16 history for out GEMM
    gru_layer_step<false><<<128, 128, 0, stream>>>(
        nullptr, hbf[wb] + BH, GRU_H, GRU_H,
        wzi + 2 * HI, wri + 2 * HI, wgi + 2 * HI,
        wzh + 2 * HI, wrh + 2 * HI, wgh + 2 * HI,
        bzh + 2 * GRU_H, brh + 2 * GRU_H, bgh + 2 * GRU_H,
        hf[rb] + 2 * BH, hbf[rb] + 2 * BH,
        hf[wb] + 2 * BH, hbf[wb] + 2 * BH,
        hist + (size_t)t * BH);
  }

  // after 512 steps (even), final state lives in buffer 0
  copy_hidden<<<(GRU_B * GRU_L * GRU_H + 255) / 256, 256, 0, stream>>>(
      hf[0], out_hid);

  // ---- batched output projection over all timesteps ----
  out_gemm<<<8192, 256, 0, stream>>>(hist, wout, bout, out_seq);
}
